// GCN_7645041787420
// MI455X (gfx1250) — matrix operand order repaired and checked by reading, not compiled, not run on hardware
//
#include <hip/hip_runtime.h>
#include <hip/hip_bf16.h>

typedef __attribute__((ext_vector_type(16))) _Float16 v16h;
typedef __attribute__((ext_vector_type(8)))  _Float16 h8v;
typedef __attribute__((ext_vector_type(4)))  _Float16 h4v;
typedef __attribute__((ext_vector_type(8)))  float    v8f;

// ---------------- utility kernels ----------------

__global__ __launch_bounds__(256) void zero_f32_kernel(float* p, int n) {
    int i = blockIdx.x * blockDim.x + threadIdx.x;
    if (i < n) p[i] = 0.0f;
}

__global__ __launch_bounds__(256) void deg_kernel(const int* __restrict__ dst,
                                                  int* __restrict__ deg, int E) {
    int e = blockIdx.x * blockDim.x + threadIdx.x;
    if (e < E) atomicAdd(&deg[dst[e]], 1);
}

__global__ __launch_bounds__(256) void dinv_kernel(const int* __restrict__ deg,
                                                   float* __restrict__ dinv, int N) {
    int i = blockIdx.x * blockDim.x + threadIdx.x;
    if (i < N) dinv[i] = rsqrtf((float)deg[i] + 1.0f);  // +1 = self loop; always > 0
}

// W1[128][64] -> Wt1[n][k] f16 ; W2[64][16] -> Wt2[n][k] f16
__global__ __launch_bounds__(256) void convw_kernel(const float* __restrict__ W1,
                                                    const float* __restrict__ W2,
                                                    _Float16* __restrict__ Wt1,
                                                    _Float16* __restrict__ Wt2) {
    int i = blockIdx.x * blockDim.x + threadIdx.x;
    if (i < 128 * 64) {
        int k = i >> 6, n = i & 63;
        Wt1[n * 128 + k] = (_Float16)W1[i];
    } else if (i < 128 * 64 + 64 * 16) {
        int j = i - 128 * 64;
        int k = j >> 4, n = j & 15;
        Wt2[n * 64 + k] = (_Float16)W2[j];
    }
}

// issue one async global->LDS 16-byte copy (CDNA5 path, tracked by ASYNCcnt)
__device__ __forceinline__ void async_copy16(unsigned ldsAddr, unsigned long long gAddr) {
    asm volatile("global_load_async_to_lds_b128 %0, %1, off"
                 :: "v"(ldsAddr), "v"(gAddr) : "memory");
}
__device__ __forceinline__ void wait_async0() {
    asm volatile("s_wait_asynccnt 0" ::: "memory");
}

// ---------------- WMMA GEMM 1: h1[N,64] = x[N,128] @ W1[128,64] ----------------
// 128 threads = 4 waves; block covers 64 rows; wave w rows [16w,16w+16), all 4 col tiles.
__global__ __launch_bounds__(128) void gemm1_kernel(const float* __restrict__ x,
                                                    const _Float16* __restrict__ Wt1,
                                                    float* __restrict__ h1, int N) {
    __shared__ __align__(32) _Float16 As[64 * 128];  // [m][k] f16, 16KB
    __shared__ __align__(32) _Float16 Bs[64 * 128];  // [n][k] f16, 16KB
    const int tid = threadIdx.x;
    const int row0 = blockIdx.x * 64;

    // async-stage B tile (f16 weights): 16384 bytes = 8 x (128 lanes x 16B)
    {
        unsigned ldsBase = (unsigned)(size_t)&Bs[0];
        unsigned long long gBase = (unsigned long long)(size_t)Wt1;
#pragma unroll
        for (int it = 0; it < 8; ++it) {
            unsigned byteOff = (unsigned)(it * 128 + tid) * 16u;
            async_copy16(ldsBase + byteOff, gBase + byteOff);
        }
    }
    // stage A tile (f32 -> f16) with float4 loads / 8B LDS stores
    {
        const float4* xv = (const float4*)x;
        h4v* As4 = (h4v*)As;
#pragma unroll
        for (int it = 0; it < 16; ++it) {
            int g = it * 128 + tid;   // 2048 float4 groups = 64 rows x 32 groups
            int m = g >> 5;
            int kg = (g & 31) << 2;
            int r = row0 + m;
            float4 f = (r < N) ? xv[((size_t)r * 128 + kg) >> 2]
                               : make_float4(0.f, 0.f, 0.f, 0.f);
            h4v h;
            h.x = (_Float16)f.x; h.y = (_Float16)f.y;
            h.z = (_Float16)f.z; h.w = (_Float16)f.w;
            As4[(m * 128 + kg) >> 2] = h;
        }
    }
    wait_async0();
    __syncthreads();

    const int wave = tid >> 5;
    const int lane = tid & 31;
    const int lane16 = lane & 15;
    const int hi = lane >> 4;      // 0: lanes 0-15, 1: lanes 16-31
    const int mrow = wave << 4;    // wave's 16-row slice

    const h8v* As8 = (const h8v*)As;
    v8f acc[4] = {{}, {}, {}, {}};

#pragma unroll
    for (int k0 = 0; k0 < 128; k0 += 32) {
        // A fragment: two contiguous 8-half runs per lane -> 2x ds_load_b128
        int abase = (mrow + lane16) * 128 + k0 + (hi << 4);
        h8v alo = As8[abase >> 3];
        h8v ahi = As8[(abase + 8) >> 3];
        v16h a = __builtin_shufflevector(alo, ahi,
                 0, 1, 2, 3, 4, 5, 6, 7, 8, 9, 10, 11, 12, 13, 14, 15);
#pragma unroll
        for (int nt = 0; nt < 4; ++nt) {
            // B fragment: one contiguous 16-half run -> 2x ds_load_b128
            const v16h b = *(const v16h*)&Bs[(nt * 16 + lane16) * 128 + k0 + (hi << 4)];
            acc[nt] = __builtin_amdgcn_wmma_f32_16x16x32_f16(false, a, false, b,
                                                             (short)0, acc[nt],
                                                             false, false);
        }
    }

    // C layout: VGPR r -> row = hi*8 + r, col = lane16
#pragma unroll
    for (int nt = 0; nt < 4; ++nt) {
#pragma unroll
        for (int r = 0; r < 8; ++r) {
            int row = row0 + mrow + (hi << 3) + r;
            if (row < N) h1[(size_t)row * 64 + nt * 16 + lane16] = acc[nt][r];
        }
    }
}

// ---------------- WMMA GEMM 2: h2[N,16] = a1[N,64] @ W2[64,16] ----------------
// 128 threads = 4 waves; block covers 64 rows; wave w rows [16w,16w+16), cols 0..15.
__global__ __launch_bounds__(128) void gemm2_kernel(const float* __restrict__ a1,
                                                    const _Float16* __restrict__ Wt2,
                                                    float* __restrict__ h2, int N) {
    __shared__ __align__(32) _Float16 As[64 * 64];  // 8KB
    __shared__ __align__(32) _Float16 Bs[16 * 64];  // 2KB
    const int tid = threadIdx.x;
    const int row0 = blockIdx.x * 64;

    // async-stage B tile: 2048 bytes = 128 lanes x 16B
    {
        unsigned ldsBase = (unsigned)(size_t)&Bs[0];
        unsigned long long gBase = (unsigned long long)(size_t)Wt2;
        unsigned byteOff = (unsigned)tid * 16u;
        async_copy16(ldsBase + byteOff, gBase + byteOff);
    }
    // stage A tile (f32 -> f16): 1024 float4 groups
    {
        const float4* av = (const float4*)a1;
        h4v* As4 = (h4v*)As;
#pragma unroll
        for (int it = 0; it < 8; ++it) {
            int g = it * 128 + tid;   // 64 rows x 16 groups
            int m = g >> 4;
            int kg = (g & 15) << 2;
            int r = row0 + m;
            float4 f = (r < N) ? av[((size_t)r * 64 + kg) >> 2]
                               : make_float4(0.f, 0.f, 0.f, 0.f);
            h4v h;
            h.x = (_Float16)f.x; h.y = (_Float16)f.y;
            h.z = (_Float16)f.z; h.w = (_Float16)f.w;
            As4[(m * 64 + kg) >> 2] = h;
        }
    }
    wait_async0();
    __syncthreads();

    const int wave = tid >> 5;
    const int lane = tid & 31;
    const int lane16 = lane & 15;
    const int hi = lane >> 4;

    const h8v* As8 = (const h8v*)As;
    v8f acc = {};

#pragma unroll
    for (int k0 = 0; k0 < 64; k0 += 32) {
        int abase = (wave * 16 + lane16) * 64 + k0 + (hi << 4);
        h8v alo = As8[abase >> 3];
        h8v ahi = As8[(abase + 8) >> 3];
        v16h a = __builtin_shufflevector(alo, ahi,
                 0, 1, 2, 3, 4, 5, 6, 7, 8, 9, 10, 11, 12, 13, 14, 15);
        const v16h b = *(const v16h*)&Bs[lane16 * 64 + k0 + (hi << 4)];
        acc = __builtin_amdgcn_wmma_f32_16x16x32_f16(false, a, false, b,
                                                     (short)0, acc, false, false);
    }

#pragma unroll
    for (int r = 0; r < 8; ++r) {
        int row = row0 + wave * 16 + (hi << 3) + r;
        if (row < N) h2[(size_t)row * 16 + lane16] = acc[r];
    }
}

// ---------------- self-loop init: out[i][c] = dinv[i]^2 * h[i][c] ----------------
__global__ __launch_bounds__(256) void selfloop_kernel(const float* __restrict__ h,
                                                       const float* __restrict__ dinv,
                                                       float* __restrict__ out,
                                                       int N, int C_log2) {
    int idx = blockIdx.x * blockDim.x + threadIdx.x;
    if (idx < (N << C_log2)) {
        int i = idx >> C_log2;
        float di = dinv[i];
        out[idx] = di * di * h[idx];
    }
}

// ---------------- edge scatter: out[dst] += dinv[s]*dinv[d] * h[src] ----------------
// lanes_per_edge lanes, each handles 4 channels via float4 + 4 f32 atomics (L2-resident).
__global__ __launch_bounds__(256) void agg_kernel(const int* __restrict__ src,
                                                  const int* __restrict__ dst,
                                                  const float* __restrict__ dinv,
                                                  const float* __restrict__ h,
                                                  float* __restrict__ out,
                                                  int E, int C, int lpe_log2) {
    long long idx = (long long)blockIdx.x * blockDim.x + threadIdx.x;
    long long total = (long long)E << lpe_log2;
    if (idx >= total) return;
    int e = (int)(idx >> lpe_log2);
    int c4 = ((int)idx & ((1 << lpe_log2) - 1)) << 2;
    int s = src[e], d = dst[e];
    float norm = dinv[s] * dinv[d];
    const float4* hp = (const float4*)(h + (size_t)s * C + c4);
    float4 m = *hp;
    float* op = out + (size_t)d * C + c4;
    atomicAdd(op + 0, norm * m.x);
    atomicAdd(op + 1, norm * m.y);
    atomicAdd(op + 2, norm * m.z);
    atomicAdd(op + 3, norm * m.w);
}

// ---------------- bias + tanh (in place) ----------------
__global__ __launch_bounds__(256) void act_kernel(float* __restrict__ out,
                                                  const float* __restrict__ bias,
                                                  int N, int C_log2) {
    int idx = blockIdx.x * blockDim.x + threadIdx.x;
    if (idx < (N << C_log2)) {
        int c = idx & ((1 << C_log2) - 1);
        out[idx] = tanhf(out[idx] + bias[c]);
    }
}

// ---------------- pooling: atomic sums + counts ----------------
__global__ __launch_bounds__(256) void pool_kernel(const float* __restrict__ h,
                                                   const int* __restrict__ batch,
                                                   float* __restrict__ psum,
                                                   float* __restrict__ pcnt, int N) {
    int idx = blockIdx.x * blockDim.x + threadIdx.x;
    if (idx < N * 16) {
        int i = idx >> 4, c = idx & 15;
        int g = batch[i];
        atomicAdd(&psum[g * 16 + c], h[idx]);
        if (c == 0) atomicAdd(&pcnt[g], 1.0f);
    }
}

__global__ __launch_bounds__(256) void final_kernel(const float* __restrict__ psum,
                                                    const float* __restrict__ pcnt,
                                                    float* __restrict__ out, int G) {
    int idx = blockIdx.x * blockDim.x + threadIdx.x;
    if (idx < G * 16) {
        int g = idx >> 4;
        float c = pcnt[g];
        float mean = psum[idx] / fmaxf(c, 1.0f);
        out[idx] = 1.0f / (1.0f + expf(-mean));
    }
}

// ---------------- launch ----------------
extern "C" void kernel_launch(void* const* d_in, const int* in_sizes, int n_in,
                              void* d_out, int out_size, void* d_ws, size_t ws_size,
                              hipStream_t stream) {
    const float* x    = (const float*)d_in[0];
    const int*   ei   = (const int*)d_in[1];
    const int*   bat  = (const int*)d_in[2];
    const float* W1   = (const float*)d_in[3];
    const float* b1   = (const float*)d_in[4];
    const float* W2   = (const float*)d_in[5];
    const float* b2   = (const float*)d_in[6];
    float* out = (float*)d_out;

    const int N = in_sizes[2];
    const int E = in_sizes[1] / 2;
    const int G = out_size / 16;
    const int* src = ei;
    const int* dst = ei + E;

    // workspace carve-out (256B aligned)
    char* ws = (char*)d_ws;
    size_t off = 0;
    auto alloc = [&](size_t bytes) {
        void* p = ws + off;
        off = (off + bytes + 255) & ~(size_t)255;
        return p;
    };
    int*       deg  = (int*)alloc((size_t)N * 4);
    float*     dinv = (float*)alloc((size_t)N * 4);
    float*     h1   = (float*)alloc((size_t)N * 64 * 4);
    float*     out1 = (float*)alloc((size_t)N * 64 * 4);   // becomes a1 after act
    float*     h2   = (float*)alloc((size_t)N * 16 * 4);
    float*     out2 = (float*)alloc((size_t)N * 16 * 4);
    _Float16*  Wt1  = (_Float16*)alloc(128 * 64 * 2);
    _Float16*  Wt2  = (_Float16*)alloc(64 * 16 * 2);
    float*     psum = (float*)alloc((size_t)G * 16 * 4);
    float*     pcnt = (float*)alloc((size_t)G * 4);
    (void)ws_size;

    const int T = 256;
    auto blocks = [&](long long n) { return (int)((n + T - 1) / T); };

    // zero accumulators
    zero_f32_kernel<<<blocks(N), T, 0, stream>>>((float*)deg, N);
    zero_f32_kernel<<<blocks(G * 16), T, 0, stream>>>(psum, G * 16);
    zero_f32_kernel<<<blocks(G), T, 0, stream>>>(pcnt, G);

    // degrees + norm
    deg_kernel<<<blocks(E), T, 0, stream>>>(dst, deg, E);
    dinv_kernel<<<blocks(N), T, 0, stream>>>(deg, dinv, N);

    // weight convert/transpose to f16
    convw_kernel<<<blocks(128 * 64 + 64 * 16), T, 0, stream>>>(W1, W2, Wt1, Wt2);

    // layer 1
    gemm1_kernel<<<(N + 63) / 64, 128, 0, stream>>>(x, Wt1, h1, N);
    selfloop_kernel<<<blocks((long long)N * 64), T, 0, stream>>>(h1, dinv, out1, N, 6);
    agg_kernel<<<blocks((long long)E * 16), T, 0, stream>>>(src, dst, dinv, h1, out1, E, 64, 4);
    act_kernel<<<blocks((long long)N * 64), T, 0, stream>>>(out1, b1, N, 6);

    // layer 2
    gemm2_kernel<<<(N + 63) / 64, 128, 0, stream>>>(out1, Wt2, h2, N);
    selfloop_kernel<<<blocks((long long)N * 16), T, 0, stream>>>(h2, dinv, out2, N, 4);
    agg_kernel<<<blocks((long long)E * 4), T, 0, stream>>>(src, dst, dinv, h2, out2, E, 16, 2);
    act_kernel<<<blocks((long long)N * 16), T, 0, stream>>>(out2, b2, N, 4);

    // pool + sigmoid
    pool_kernel<<<blocks((long long)N * 16), T, 0, stream>>>(out2, bat, psum, pcnt, N);
    final_kernel<<<blocks(G * 16), T, 0, stream>>>(psum, pcnt, out, G);
}
